// ALiBiMultiheadAttention_57372173140155
// MI455X (gfx1250) — compile-verified, matrix operand order checked by reading, not run on hardware
//
#include <hip/hip_runtime.h>
#include <hip/hip_bf16.h>

// ---------------- problem constants (match reference) ----------------
#define D_MODEL 1024
#define NHEAD   16
#define DH      64
#define SEQ     2048
#define BATCH   2
#define PBLEN   (2*SEQ - 1)   // 4095 (MAX_LEN == SEQ here)
#define MROWS   (BATCH*SEQ)   // 4096

typedef __bf16 bf16_t;
typedef float  v8f   __attribute__((ext_vector_type(8)));
typedef __bf16 v16bf __attribute__((ext_vector_type(16)));
typedef __bf16 v8bf  __attribute__((ext_vector_type(8)));

union BF16x16 { v16bf v; v8bf h[2]; };

// Load one 16x32 bf16 operand tile in the CDNA5 WMMA A/B lane layout:
// lane L holds row (L&15); its 16 elements are K = base..base+7 and
// base+16..base+23 with base = 8*(L>>4).  `p` points at element
// (lane_row, k0 + base); the second b128 chunk is +16 elements.
__device__ __forceinline__ v16bf ld_ab_tile(const bf16_t* p) {
  BF16x16 r;
  r.h[0] = *(const v8bf*)(p);
  r.h[1] = *(const v8bf*)(p + 16);
  return r.v;
}

__device__ __forceinline__ v8f vzero8() {
  v8f z = {0.f,0.f,0.f,0.f,0.f,0.f,0.f,0.f};
  return z;
}

__device__ __forceinline__ v8f wmma_bf16(v16bf a, v16bf b, v8f c) {
  return __builtin_amdgcn_wmma_f32_16x16x32_bf16(
      /*neg_a=*/false, a, /*neg_b=*/false, b,
      /*c_mod=*/(short)0, c, /*reuse_a=*/false, /*reuse_b=*/false);
}

// ---------------- fp32 -> bf16 convert ----------------
__global__ void f32_to_bf16_kernel(const float* __restrict__ in,
                                   bf16_t* __restrict__ out, size_t n) {
  size_t i = (size_t)blockIdx.x * blockDim.x + threadIdx.x;
  if (i < n) out[i] = (bf16_t)in[i];
}

// ---------------- shared GEMM mainloop -------------------------------
// C[M,N] = A[M,K] * W[N,K]^T.  Wave computes a 32x64 tile, 8 WMMAs per
// K-step of 32.  Bases are wave-uniform (SGPR), lane offsets 32-bit.
__device__ __forceinline__ void gemm_core(const bf16_t* __restrict__ A,
                                          const bf16_t* __restrict__ W,
                                          int K, int m0, int n0,
                                          int l15, int kb8,
                                          v8f acc[2][4]) {
#pragma unroll
  for (int t = 0; t < 2; ++t)
#pragma unroll
    for (int u = 0; u < 4; ++u) acc[t][u] = vzero8();

  const bf16_t* __restrict__ Abase = A + (size_t)m0 * K;  // uniform
  const bf16_t* __restrict__ Wbase = W + (size_t)n0 * K;  // uniform
  int aoff[2], woff[4];
#pragma unroll
  for (int t = 0; t < 2; ++t) aoff[t] = (t*16 + l15) * K + kb8;
#pragma unroll
  for (int u = 0; u < 4; ++u) woff[u] = (u*16 + l15) * K + kb8;

  for (int k0 = 0; k0 < K; k0 += 32) {
    v16bf a[2], b[4];
#pragma unroll
    for (int t = 0; t < 2; ++t) a[t] = ld_ab_tile(Abase + aoff[t] + k0);
#pragma unroll
    for (int u = 0; u < 4; ++u) b[u] = ld_ab_tile(Wbase + woff[u] + k0);
#pragma unroll
    for (int t = 0; t < 2; ++t)
#pragma unroll
      for (int u = 0; u < 4; ++u)
        acc[t][u] = wmma_bf16(a[t], b[u], acc[t][u]);
  }
}

// ---------------- GEMM, bf16 output ----------------
__global__ __launch_bounds__(256)
void gemm_bf16_out(const bf16_t* __restrict__ A,
                   const bf16_t* __restrict__ W,
                   const float*  __restrict__ bias,
                   bf16_t* __restrict__ C,
                   int M, int N, int K) {
  const int lane = threadIdx.x & 31;
  const int wave = threadIdx.x >> 5;
  const int m0 = blockIdx.x * 128 + (wave >> 1) * 32;
  const int n0 = blockIdx.y * 128 + (wave & 1) * 64;
  const int l15 = lane & 15;
  const int kh  = lane >> 4;
  const int kb8 = kh * 8;

  v8f acc[2][4];
  gemm_core(A, W, K, m0, n0, l15, kb8, acc);

  bf16_t* __restrict__ Cbase = C + (size_t)m0 * N + n0;   // uniform
#pragma unroll
  for (int u = 0; u < 4; ++u) {
    const int nc = u*16 + l15;
    const float bv = bias[n0 + nc];
#pragma unroll
    for (int t = 0; t < 2; ++t) {
#pragma unroll
      for (int i = 0; i < 8; ++i) {
        const int mr = t*16 + i + 8*kh;
        Cbase[(size_t)mr * N + nc] = (bf16_t)(acc[t][u][i] + bv);
      }
    }
  }
}

// ---------------- GEMM, fp32 output ----------------
__global__ __launch_bounds__(256)
void gemm_f32_out(const bf16_t* __restrict__ A,
                  const bf16_t* __restrict__ W,
                  const float*  __restrict__ bias,
                  float* __restrict__ C,
                  int M, int N, int K) {
  const int lane = threadIdx.x & 31;
  const int wave = threadIdx.x >> 5;
  const int m0 = blockIdx.x * 128 + (wave >> 1) * 32;
  const int n0 = blockIdx.y * 128 + (wave & 1) * 64;
  const int l15 = lane & 15;
  const int kh  = lane >> 4;
  const int kb8 = kh * 8;

  v8f acc[2][4];
  gemm_core(A, W, K, m0, n0, l15, kb8, acc);

  float* __restrict__ Cbase = C + (size_t)m0 * N + n0;    // uniform
#pragma unroll
  for (int u = 0; u < 4; ++u) {
    const int nc = u*16 + l15;
    const float bv = bias[n0 + nc];
#pragma unroll
    for (int t = 0; t < 2; ++t) {
#pragma unroll
      for (int i = 0; i < 8; ++i) {
        const int mr = t*16 + i + 8*kh;
        Cbase[(size_t)mr * N + nc] = acc[t][u][i] + bv;
      }
    }
  }
}

// ---------------- V transpose: [B*L, H*DH] -> [B*H, DH, L] ----------------
__global__ void transpose_v_kernel(const bf16_t* __restrict__ V,
                                   bf16_t* __restrict__ Vt) {
  size_t idx = (size_t)blockIdx.x * blockDim.x + threadIdx.x;
  const size_t total = (size_t)BATCH * NHEAD * DH * SEQ;
  if (idx >= total) return;
  const int l  = (int)(idx % SEQ);
  size_t r = idx / SEQ;
  const int d  = (int)(r % DH);  r /= DH;
  const int h  = (int)(r % NHEAD);
  const int b  = (int)(r / NHEAD);
  Vt[idx] = V[((size_t)b * SEQ + l) * D_MODEL + h * DH + d];
}

// ---------------- flash attention (one wave per 16-row Q block) ----------
__global__ __launch_bounds__(128)
void attn_flash_kernel(const bf16_t* __restrict__ Q,
                       const bf16_t* __restrict__ Kk,
                       const bf16_t* __restrict__ Vt,
                       const float*  __restrict__ pos_bias,
                       bf16_t* __restrict__ Ctx) {
  __shared__ float pb[PBLEN];
  __shared__ __align__(16) bf16_t pbuf[4][16 * 32];

  const int h = blockIdx.y;
  const int b = blockIdx.z;

  // stage this head's positional-bias row into LDS (shared by all 4 waves)
  for (int i = threadIdx.x; i < PBLEN; i += 128)
    pb[i] = pos_bias[(size_t)h * PBLEN + i];
  __syncthreads();

  const int lane = threadIdx.x & 31;
  const int wave = threadIdx.x >> 5;
  const int q0   = (blockIdx.x * 4 + wave) * 16;
  const int l15  = lane & 15;
  const int kh   = lane >> 4;
  const int kb8  = kh * 8;

  // wave-uniform bases (SGPR), 32-bit lane offsets
  const bf16_t* __restrict__ Qb = Q  + ((size_t)b * SEQ + q0) * D_MODEL + h * DH;
  const bf16_t* __restrict__ Kb = Kk + ((size_t)b * SEQ) * D_MODEL + h * DH;
  const bf16_t* __restrict__ Vb = Vt + ((size_t)(b * NHEAD + h) * DH) * (size_t)SEQ;

  const int qoff = l15 * D_MODEL + kb8;
  const int koff = l15 * D_MODEL + kb8;
  const int voff = l15 * SEQ + kb8;

  // Q tile (16x64) stays in registers: two K-steps of 32
  v16bf aq0 = ld_ab_tile(Qb + qoff + 0);
  v16bf aq1 = ld_ab_tile(Qb + qoff + 32);

  v8f acc[4];
#pragma unroll
  for (int u = 0; u < 4; ++u) acc[u] = vzero8();
  float mrow[8], lrow[8];
#pragma unroll
  for (int i = 0; i < 8; ++i) { mrow[i] = -1e30f; lrow[i] = 0.f; }

  const float scale = 0.125f; // 1/sqrt(64)

  for (int kb = 0; kb < SEQ; kb += 32) {
    // ---- S = Q * K^T  (16 x 32), fp32 accum ----
    v8f s[2];
#pragma unroll
    for (int n = 0; n < 2; ++n) {
      const bf16_t* kp = Kb + (size_t)(kb + n*16) * D_MODEL;
      v16bf bk0 = ld_ab_tile(kp + koff + 0);
      v16bf bk1 = ld_ab_tile(kp + koff + 32);
      v8f t = vzero8();
      t = wmma_bf16(aq0, bk0, t);
      t = wmma_bf16(aq1, bk1, t);
      s[n] = t;
    }

    // ---- scale + ALiBi bias (LDS) ----
#pragma unroll
    for (int i = 0; i < 8; ++i) {
      const int q = q0 + i + 8 * kh;
      const int base = q + (SEQ - 1) - kb;        // idx = q - k + 2047
      s[0][i] = s[0][i] * scale + pb[base - l15];
      s[1][i] = s[1][i] * scale + pb[base - 16 - l15];
    }

    // ---- row max (16 lanes per half share a row) + rescale factors ----
    float alpha[8], mnew[8];
#pragma unroll
    for (int i = 0; i < 8; ++i) {
      float v = fmaxf(s[0][i], s[1][i]);
      v = fmaxf(v, __shfl_xor(v, 1));
      v = fmaxf(v, __shfl_xor(v, 2));
      v = fmaxf(v, __shfl_xor(v, 4));
      v = fmaxf(v, __shfl_xor(v, 8));
      mnew[i]  = fmaxf(mrow[i], v);
      alpha[i] = __expf(mrow[i] - mnew[i]);
      mrow[i]  = mnew[i];
    }

    // ---- P = exp(S - m), update running sums ----
#pragma unroll
    for (int i = 0; i < 8; ++i) {
      float p0 = __expf(s[0][i] - mnew[i]);
      float p1 = __expf(s[1][i] - mnew[i]);
      s[0][i] = p0; s[1][i] = p1;
      float ps = p0 + p1;
      ps += __shfl_xor(ps, 1);
      ps += __shfl_xor(ps, 2);
      ps += __shfl_xor(ps, 4);
      ps += __shfl_xor(ps, 8);
      lrow[i] = lrow[i] * alpha[i] + ps;
    }

    // ---- rescale output accumulators ----
#pragma unroll
    for (int u = 0; u < 4; ++u)
#pragma unroll
      for (int i = 0; i < 8; ++i)
        acc[u][i] *= alpha[i];

    // ---- C-layout -> A-layout reswizzle of P through LDS ----
#pragma unroll
    for (int i = 0; i < 8; ++i) {
      const int m = i + 8 * kh;
      pbuf[wave][m * 32 + l15]      = (bf16_t)s[0][i];
      pbuf[wave][m * 32 + 16 + l15] = (bf16_t)s[1][i];
    }
    __syncthreads();
    BF16x16 ap;
    ap.h[0] = *(const v8bf*)&pbuf[wave][l15 * 32 + kb8];
    ap.h[1] = *(const v8bf*)&pbuf[wave][l15 * 32 + kb8 + 16];
    __syncthreads();

    // ---- O += P (16x32) * V (32x64), V pre-transposed so loads contiguous
#pragma unroll
    for (int u = 0; u < 4; ++u) {
      v16bf bv = ld_ab_tile(Vb + (size_t)(u*16) * SEQ + voff + kb);
      acc[u] = wmma_bf16(ap.v, bv, acc[u]);
    }
  }

  // ---- normalize and write context (bf16) ----
#pragma unroll
  for (int i = 0; i < 8; ++i) lrow[i] = 1.0f / lrow[i];
  bf16_t* __restrict__ Cbase = Ctx + ((size_t)b * SEQ + q0) * D_MODEL + h * DH;
#pragma unroll
  for (int u = 0; u < 4; ++u) {
#pragma unroll
    for (int i = 0; i < 8; ++i) {
      const int mr = i + 8 * kh;
      const int nc = u * 16 + l15;
      Cbase[(size_t)mr * D_MODEL + nc] = (bf16_t)(acc[u][i] * lrow[i]);
    }
  }
}

// ---------------- host side ----------------
extern "C" void kernel_launch(void* const* d_in, const int* in_sizes, int n_in,
                              void* d_out, int out_size, void* d_ws, size_t ws_size,
                              hipStream_t stream) {
  (void)in_sizes; (void)n_in; (void)out_size; (void)ws_size;

  const float* queries  = (const float*)d_in[0];
  const float* Wq       = (const float*)d_in[1];
  const float* bq       = (const float*)d_in[2];
  const float* Wk       = (const float*)d_in[3];
  const float* bk       = (const float*)d_in[4];
  const float* Wv       = (const float*)d_in[5];
  const float* bv       = (const float*)d_in[6];
  const float* Wo       = (const float*)d_in[7];
  const float* bo       = (const float*)d_in[8];
  const float* pos_bias = (const float*)d_in[9];
  float* out = (float*)d_out;

  const size_t MK = (size_t)MROWS * D_MODEL;   // 4,194,304 activation elems
  const size_t WK = (size_t)D_MODEL * D_MODEL; // 1,048,576 weight elems

  bf16_t* p   = (bf16_t*)d_ws;
  bf16_t* Xb  = p; p += MK;
  bf16_t* Wqb = p; p += WK;
  bf16_t* Wkb = p; p += WK;
  bf16_t* Wvb = p; p += WK;
  bf16_t* Wob = p; p += WK;
  bf16_t* Qb  = p; p += MK;
  bf16_t* Kb  = p; p += MK;
  bf16_t* Vb  = p; p += MK;
  bf16_t* Vt  = p; p += MK;
  bf16_t* Ctx = p; p += MK;

  // 1) fp32 -> bf16 staging
  f32_to_bf16_kernel<<<(unsigned)((MK + 255) / 256), 256, 0, stream>>>(queries, Xb, MK);
  f32_to_bf16_kernel<<<(unsigned)((WK + 255) / 256), 256, 0, stream>>>(Wq, Wqb, WK);
  f32_to_bf16_kernel<<<(unsigned)((WK + 255) / 256), 256, 0, stream>>>(Wk, Wkb, WK);
  f32_to_bf16_kernel<<<(unsigned)((WK + 255) / 256), 256, 0, stream>>>(Wv, Wvb, WK);
  f32_to_bf16_kernel<<<(unsigned)((WK + 255) / 256), 256, 0, stream>>>(Wo, Wob, WK);

  // 2) QKV projections (WMMA GEMM, bf16 out)
  dim3 gg(MROWS / 128, D_MODEL / 128);
  gemm_bf16_out<<<gg, 256, 0, stream>>>(Xb, Wqb, bq, Qb, MROWS, D_MODEL, D_MODEL);
  gemm_bf16_out<<<gg, 256, 0, stream>>>(Xb, Wkb, bk, Kb, MROWS, D_MODEL, D_MODEL);
  gemm_bf16_out<<<gg, 256, 0, stream>>>(Xb, Wvb, bv, Vb, MROWS, D_MODEL, D_MODEL);

  // 3) V transpose for contiguous B-operand loads in attention
  transpose_v_kernel<<<(unsigned)((MK + 255) / 256), 256, 0, stream>>>(Vb, Vt);

  // 4) flash attention (WMMA QK^T + PV), bf16 context out
  attn_flash_kernel<<<dim3(SEQ / 64, NHEAD, BATCH), 128, 0, stream>>>(Qb, Kb, Vt, pos_bias, Ctx);

  // 5) output projection (WMMA GEMM, fp32 out to d_out)
  gemm_f32_out<<<gg, 256, 0, stream>>>(Ctx, Wob, bo, out, MROWS, D_MODEL, D_MODEL);
}
